// GRUWithAttention_50173807952558
// MI455X (gfx1250) — compile-verified, matrix operand order checked by reading, not run on hardware
//
#include <hip/hip_runtime.h>
#include <hip/hip_bf16.h>

// ---------------------------------------------------------------------------
// Problem constants
// ---------------------------------------------------------------------------
constexpr int B = 128;
constexpr int T = 512;
constexpr int E = 256;
constexpr int H = 512;
constexpr int A = 64;

typedef __attribute__((ext_vector_type(16))) __bf16 v16bf;
typedef __attribute__((ext_vector_type(8)))  __bf16 v8bf;
typedef __attribute__((ext_vector_type(8)))  float  v8f;

// ---------------------------------------------------------------------------
// Helpers
// ---------------------------------------------------------------------------
__device__ __forceinline__ __bf16 f32_to_bf16(float f) {
  unsigned u = __builtin_bit_cast(unsigned, f);
  u += 0x7FFFu + ((u >> 16) & 1u);                 // round-to-nearest-even
  unsigned short h = (unsigned short)(u >> 16);
  return __builtin_bit_cast(__bf16, h);
}

__device__ __forceinline__ float sigmoidf_(float x) {
  return 1.0f / (1.0f + __expf(-x));
}

// Fast tanh: prefer the gfx1250 TRANS-unit v_tanh_f32; else one v_exp_f32.
__device__ __forceinline__ float fast_tanhf(float x) {
#if __has_builtin(__builtin_amdgcn_tanhf)
  return __builtin_amdgcn_tanhf(x);
#else
  return 1.0f - 2.0f / (__expf(2.0f * x) + 1.0f);
#endif
}

// WMMA bf16: D(16x16 f32) = A(16x32 bf16) * B(32x16 bf16) + C
__device__ __forceinline__ v8f wmma_bf16(v16bf a, v16bf b, v8f c) {
  return __builtin_amdgcn_wmma_f32_16x16x32_bf16(false, a, false, b,
                                                 (short)0, c, false, false);
}

// Load a 16x32 bf16 tile (rows row0..row0+15, cols k0..k0+31) from a
// row-major matrix with leading dim ld, into the per-lane A/B fragment
// layout (ISA 7.12.2):
//   lane L (0-15):  row = L, holds K = k0+0..7  and k0+16..23
//   lane L+16:      row = L, holds K = k0+8..15 and k0+24..31
__device__ __forceinline__ v16bf load_tile_bf16(const __bf16* mat, int ld,
                                                int row0, int k0, int lane) {
  int m    = lane & 15;
  int koff = (lane >> 4) << 3;   // 0 or 8
  const __bf16* p = mat + (size_t)(row0 + m) * ld + k0 + koff;
  v8bf lo = *(const v8bf*)(p);
  v8bf hi = *(const v8bf*)(p + 16);
  return __builtin_shufflevector(lo, hi, 0, 1, 2, 3, 4, 5, 6, 7,
                                         8, 9, 10, 11, 12, 13, 14, 15);
}

// Same fragment but gathered from an fp32 embedding table (row = token id).
__device__ __forceinline__ v16bf load_emb_frag(const float* emb,
                                               const int* tok_ids,
                                               int row0, int k0, int lane) {
  int m   = lane & 15;
  int tok = tok_ids[row0 + m];
  int koff = (lane >> 4) << 3;
  const float* p = emb + (size_t)tok * E + k0 + koff;
  v16bf a;
#pragma unroll
  for (int e = 0; e < 16; ++e) {
    int k = ((e & 8) << 1) + (e & 7);   // 0..7 -> 0..7 ; 8..15 -> 16..23
    a[e] = f32_to_bf16(p[k]);
  }
  return a;
}

// ---------------------------------------------------------------------------
// Prep kernels
// ---------------------------------------------------------------------------
__global__ void cvt_f32_to_bf16_kernel(const float* __restrict__ src,
                                       __bf16* __restrict__ dst, int n) {
  int i = blockIdx.x * 256 + threadIdx.x;
  if (i < n) dst[i] = f32_to_bf16(src[i]);
}

__global__ void zero_f32_kernel(float* __restrict__ p, int n) {
  int i = blockIdx.x * 256 + threadIdx.x;
  if (i < n) p[i] = 0.0f;
}

// ---------------------------------------------------------------------------
// tokens_linear:  tp[b,t,a] = sum_e emb[tok[b,t],e] * Wt[a,e] + bt[a]
// One wave per 16-row M tile of the flattened (B*T, E) activation; 4 N-tiles.
// ---------------------------------------------------------------------------
__global__ void __launch_bounds__(32)
tp_kernel(const int* __restrict__ tok, const float* __restrict__ emb,
          const __bf16* __restrict__ WtB, const float* __restrict__ bt,
          float* __restrict__ tp) {
  int mt   = blockIdx.x;           // 0 .. B*T/16 - 1
  int lane = threadIdx.x;
  int n_   = lane & 15;
  int mhi  = (lane >> 4) << 3;

  v8f acc[4];
#pragma unroll
  for (int nt = 0; nt < 4; ++nt) {
    float bias = bt[nt * 16 + n_];
#pragma unroll
    for (int r = 0; r < 8; ++r) acc[nt][r] = bias;
  }

  for (int k0 = 0; k0 < E; k0 += 32) {
    v16bf a = load_emb_frag(emb, tok, mt * 16, k0, lane);
#pragma unroll
    for (int nt = 0; nt < 4; ++nt) {
      v16bf bfrag = load_tile_bf16(WtB, E, nt * 16, k0, lane);
      acc[nt] = wmma_bf16(a, bfrag, acc[nt]);
    }
  }

#pragma unroll
  for (int nt = 0; nt < 4; ++nt) {
#pragma unroll
    for (int r = 0; r < 8; ++r) {
      int row = mt * 16 + r + mhi;
      tp[(size_t)row * A + nt * 16 + n_] = acc[nt][r];
    }
  }
}

// ---------------------------------------------------------------------------
// Attention (per step t): q = h1@Wq.T+bq ; s = tanh(tp+q)@v+bv over prefix;
// softmax; ctx = attn @ tp.  One block (256 threads) per batch row.
// ---------------------------------------------------------------------------
__global__ void __launch_bounds__(256)
attn_kernel(const float* __restrict__ tp, const float* __restrict__ h1,
            const float* __restrict__ Wq, const float* __restrict__ bq,
            const float* __restrict__ v, const float* __restrict__ bv,
            float* __restrict__ ctxf, __bf16* __restrict__ ctxbf, int t) {
  int b   = blockIdx.x;
  int tid = threadIdx.x;
  int a   = tid & 63;
  int g   = tid >> 6;              // 0..3

  __shared__ float q_s[A];
  __shared__ float s_s[T];
  __shared__ float red[256];
  __shared__ float part[4][A];

  // ---- q[b,:] : 4 partial dots of length 128 per output -------------------
  {
    const float* hseg = h1 + (size_t)b * H + g * 128;
    const float* wseg = Wq + (size_t)a * H + g * 128;
    float sum = 0.0f;
#pragma unroll 4
    for (int k = 0; k < 128; ++k) sum += hseg[k] * wseg[k];
    part[g][a] = sum;
  }
  __syncthreads();
  if (tid < A) q_s[tid] = part[0][tid] + part[1][tid] + part[2][tid] +
                          part[3][tid] + bq[tid];
  __syncthreads();

  // ---- scores over prefix 0..t -------------------------------------------
  float lmax = -INFINITY;
  float bvv  = bv[0];
  for (int tt = tid; tt <= t; tt += 256) {
    const float* trow = tp + ((size_t)b * T + tt) * A;
    float s = bvv;
#pragma unroll 4
    for (int aa = 0; aa < A; ++aa)
      s += v[aa] * fast_tanhf(trow[aa] + q_s[aa]);
    s_s[tt] = s;
    lmax = fmaxf(lmax, s);
  }
  red[tid] = lmax;
  __syncthreads();
  for (int st = 128; st > 0; st >>= 1) {
    if (tid < st) red[tid] = fmaxf(red[tid], red[tid + st]);
    __syncthreads();
  }
  float mx = red[0];
  __syncthreads();

  // ---- exp + sum ----------------------------------------------------------
  float lsum = 0.0f;
  for (int tt = tid; tt <= t; tt += 256) {
    float e = __expf(s_s[tt] - mx);
    s_s[tt] = e;
    lsum += e;
  }
  red[tid] = lsum;
  __syncthreads();
  for (int st = 128; st > 0; st >>= 1) {
    if (tid < st) red[tid] += red[tid + st];
    __syncthreads();
  }
  float denom = red[0];
  __syncthreads();

  // ---- ctx[b,a] = sum_t attn * tp ----------------------------------------
  float acc = 0.0f;
  for (int tt = g; tt <= t; tt += 4)
    acc += s_s[tt] * tp[((size_t)b * T + tt) * A + a];
  part[g][a] = acc;
  __syncthreads();
  if (tid < A) {
    float c = (part[0][tid] + part[1][tid] + part[2][tid] + part[3][tid]) /
              denom;
    ctxf[b * A + tid]  = c;
    ctxbf[b * A + tid] = f32_to_bf16(c);
  }
}

// ---------------------------------------------------------------------------
// Fused GRU cell: one wave per 32(M)x16(N) slab of h_new (two M-tiles share
// every weight fragment -> 12 WMMA accumulator tiles per wave).
//   gi = x @ Wih.T + bih (K = Kx),  gh = h_prev @ Whh.T + bhh (K = H)
//   r,z,n gates fused in-register.  Optionally writes d_out rows where
//   clip(len-2) == t.
// ---------------------------------------------------------------------------
__global__ void __launch_bounds__(32)
gru_cell_kernel(const __bf16* __restrict__ xbf, int Kx,
                const __bf16* __restrict__ Wih, const float* __restrict__ bih,
                const __bf16* __restrict__ hbf,
                const __bf16* __restrict__ Whh, const float* __restrict__ bhh,
                const float* __restrict__ h_old, float* __restrict__ h_new,
                __bf16* __restrict__ h_new_bf,
                const int* __restrict__ lengths, float* __restrict__ out,
                int t) {
  int lane = threadIdx.x;
  int pair = blockIdx.x >> 5;      // 0..3   (B/32)
  int nt   = blockIdx.x & 31;      // 0..31  (H/16)
  int n_   = lane & 15;
  int col  = nt * 16 + n_;

  // gi[m][g], gh[m][g]: m = M-tile (0/1), g = gate (0=r,1=z,2=n)
  v8f gi[2][3], gh[2][3];
  {
    float bi[3] = {bih[col], bih[H + col], bih[2 * H + col]};
    float bh[3] = {bhh[col], bhh[H + col], bhh[2 * H + col]};
#pragma unroll
    for (int m = 0; m < 2; ++m)
#pragma unroll
      for (int g = 0; g < 3; ++g)
#pragma unroll
        for (int r = 0; r < 8; ++r) {
          gi[m][g][r] = bi[g];
          gh[m][g][r] = bh[g];
        }
  }

  int mrow0 = pair * 32;           // first row of the 32-row slab

  // input-to-hidden (K = Kx)
  for (int k0 = 0; k0 < Kx; k0 += 32) {
    v16bf a0 = load_tile_bf16(xbf, Kx, mrow0,      k0, lane);
    v16bf a1 = load_tile_bf16(xbf, Kx, mrow0 + 16, k0, lane);
#pragma unroll
    for (int g = 0; g < 3; ++g) {
      v16bf bfrag = load_tile_bf16(Wih, Kx, g * H + nt * 16, k0, lane);
      gi[0][g] = wmma_bf16(a0, bfrag, gi[0][g]);
      gi[1][g] = wmma_bf16(a1, bfrag, gi[1][g]);
    }
  }
  // hidden-to-hidden (K = H)
  for (int k0 = 0; k0 < H; k0 += 32) {
    v16bf a0 = load_tile_bf16(hbf, H, mrow0,      k0, lane);
    v16bf a1 = load_tile_bf16(hbf, H, mrow0 + 16, k0, lane);
#pragma unroll
    for (int g = 0; g < 3; ++g) {
      v16bf bfrag = load_tile_bf16(Whh, H, g * H + nt * 16, k0, lane);
      gh[0][g] = wmma_bf16(a0, bfrag, gh[0][g]);
      gh[1][g] = wmma_bf16(a1, bfrag, gh[1][g]);
    }
  }

  int mhi = (lane >> 4) << 3;
#pragma unroll
  for (int m = 0; m < 2; ++m) {
#pragma unroll
    for (int r = 0; r < 8; ++r) {
      int row  = mrow0 + m * 16 + r + mhi;
      float rg = sigmoidf_(gi[m][0][r] + gh[m][0][r]);
      float zg = sigmoidf_(gi[m][1][r] + gh[m][1][r]);
      float ng = fast_tanhf(gi[m][2][r] + rg * gh[m][2][r]);
      float hv = (1.0f - zg) * ng + zg * h_old[(size_t)row * H + col];
      h_new[(size_t)row * H + col]    = hv;
      h_new_bf[(size_t)row * H + col] = f32_to_bf16(hv);
      if (out != nullptr) {
        int idx = lengths[row] - 2;
        idx = idx < 0 ? 0 : (idx > T - 1 ? T - 1 : idx);
        if (idx == t) out[(size_t)row * H + col] = hv;
      }
    }
  }
}

// ---------------------------------------------------------------------------
// Host launcher
// ---------------------------------------------------------------------------
extern "C" void kernel_launch(void* const* d_in, const int* in_sizes, int n_in,
                              void* d_out, int out_size, void* d_ws,
                              size_t ws_size, hipStream_t stream) {
  (void)in_sizes; (void)n_in; (void)out_size; (void)ws_size;

  const int*   tok     = (const int*)d_in[0];
  const int*   lengths = (const int*)d_in[1];
  const float* emb     = (const float*)d_in[2];
  const float* Wt      = (const float*)d_in[3];
  const float* bt      = (const float*)d_in[4];
  const float* Wq      = (const float*)d_in[5];
  const float* bq      = (const float*)d_in[6];
  const float* vw      = (const float*)d_in[7];
  const float* bv      = (const float*)d_in[8];
  const float* Wih0    = (const float*)d_in[9];
  const float* Whh0    = (const float*)d_in[10];
  const float* bih0    = (const float*)d_in[11];
  const float* bhh0    = (const float*)d_in[12];
  const float* Wih1    = (const float*)d_in[13];
  const float* Whh1    = (const float*)d_in[14];
  const float* bih1    = (const float*)d_in[15];
  const float* bhh1    = (const float*)d_in[16];
  float* out = (float*)d_out;

  // ---- workspace carving (all offsets 256B aligned) -----------------------
  char* w = (char*)d_ws;
  size_t off = 0;
  auto carve = [&](size_t bytes) {
    void* p = w + off;
    off += (bytes + 255) & ~(size_t)255;
    return p;
  };
  float*  tp      = (float*)carve((size_t)B * T * A * 4);   // 16 MB
  float*  hblock  = (float*)carve(0);                       // marker
  float*  h0f     = (float*)carve((size_t)2 * B * H * 4);
  float*  h1f     = (float*)carve((size_t)2 * B * H * 4);
  __bf16* h0bf    = (__bf16*)carve((size_t)2 * B * H * 2);
  __bf16* h1bf    = (__bf16*)carve((size_t)2 * B * H * 2);
  size_t  hfloats = ((size_t)2 * B * H * 4 * 2 + (size_t)2 * B * H * 2 * 2) / 4;
  float*  ctxf    = (float*)carve((size_t)B * A * 4);
  __bf16* ctxbf   = (__bf16*)carve((size_t)B * A * 2);
  __bf16* WtB     = (__bf16*)carve((size_t)A * E * 2);
  __bf16* Wih0B   = (__bf16*)carve((size_t)3 * H * A * 2);
  __bf16* Whh0B   = (__bf16*)carve((size_t)3 * H * H * 2);
  __bf16* Wih1B   = (__bf16*)carve((size_t)3 * H * H * 2);
  __bf16* Whh1B   = (__bf16*)carve((size_t)3 * H * H * 2);

  // ---- weight conversion f32 -> bf16 -------------------------------------
  auto cvt = [&](const float* s, __bf16* d, int n) {
    cvt_f32_to_bf16_kernel<<<(n + 255) / 256, 256, 0, stream>>>(s, d, n);
  };
  cvt(Wt,   WtB,   A * E);
  cvt(Wih0, Wih0B, 3 * H * A);
  cvt(Whh0, Whh0B, 3 * H * H);
  cvt(Wih1, Wih1B, 3 * H * H);
  cvt(Whh1, Whh1B, 3 * H * H);

  // ---- zero hidden-state block (both ping-pong copies) --------------------
  zero_f32_kernel<<<((int)hfloats + 255) / 256, 256, 0, stream>>>(hblock,
                                                                  (int)hfloats);

  // ---- tokens_linear via WMMA ---------------------------------------------
  tp_kernel<<<(B * T) / 16, 32, 0, stream>>>(tok, emb, WtB, bt, tp);

  // ---- sequential scan ----------------------------------------------------
  const int NH = B * H;
  const int cellGrid = (B / 32) * (H / 16);   // 128 waves
  for (int t = 0; t < T; ++t) {
    int cur = t & 1, nxt = cur ^ 1;
    attn_kernel<<<B, 256, 0, stream>>>(tp, h1f + (size_t)cur * NH, Wq, bq, vw,
                                       bv, ctxf, ctxbf, t);
    gru_cell_kernel<<<cellGrid, 32, 0, stream>>>(
        ctxbf, A, Wih0B, bih0, h0bf + (size_t)cur * NH, Whh0B, bhh0,
        h0f + (size_t)cur * NH, h0f + (size_t)nxt * NH,
        h0bf + (size_t)nxt * NH, nullptr, nullptr, t);
    gru_cell_kernel<<<cellGrid, 32, 0, stream>>>(
        h0bf + (size_t)nxt * NH, H, Wih1B, bih1, h1bf + (size_t)cur * NH,
        Whh1B, bhh1, h1f + (size_t)cur * NH, h1f + (size_t)nxt * NH,
        h1bf + (size_t)nxt * NH, lengths, out, t);
  }
}
